// RealNVP_41798621724923
// MI455X (gfx1250) — matrix operand order repaired and checked by reading, not compiled, not run on hardware
//
#include <hip/hip_runtime.h>
#include <math.h>

typedef __attribute__((ext_vector_type(16))) __bf16 v16bf;
typedef __attribute__((ext_vector_type(8)))  float  v8f;

#define LAYERS 6
#define DD 64
#define dd 32
#define HH 512
#define NN 65536
#define EPS 1e-5f
#define TM 64   // rows per block (4 x 16-row WMMA tiles)

#define W1_SZ (HH*dd)
#define B1_SZ (HH)
#define W2_SZ (HH*HH)
#define B2_SZ (HH)
#define W3_SZ (dd*HH)
#define B3_SZ (dd)

// ---------------- fragment helpers (wave32 WMMA layouts per CDNA5 ISA) -----

union AFrag { v16bf v; unsigned u[8]; uint4 q[2]; };
union BFrag { v16bf v; unsigned u[8]; uint4 q[2]; };
union CFrag { v8f  v; float    f[8]; };

// 16x32 bf16 A fragment from row-major LDS (stride in elements).
// ISA: lanes0-15 M=lane, lanes16-31 M=lane-16.
// VGPR v<4 : K = 2v,2v+1 (+8 upper half); v>=4 : K = 16+2(v-4),.. (+8 upper).
// => per lane: 8 contiguous bf16 at k=half*8 and 8 more at k=16+half*8.
__device__ inline v16bf load_a_lds16(const __bf16* base, int stride, int lane) {
  int m = lane & 15, half = lane >> 4;
  const __bf16* p = base + m * stride + half * 8;
  AFrag a;
  a.q[0] = *(const uint4*)(p);        // ds_load_b128
  a.q[1] = *(const uint4*)(p + 16);   // ds_load_b128
  return a.v;
}

// 32x16 bf16 B fragment for output cols [n0,n0+16), K chunk base kb.
// W row-major [rows x Kdim]; B(k,n) = W[n0+n][kb+k].
// per lane: 8 contiguous bf16 at k=half*8 and 8 more at k=16+half*8 (matches A).
__device__ inline v16bf load_b_w(const __bf16* __restrict__ W, int Kdim,
                                 int n0, int kb, int lane) {
  int n = lane & 15, half = lane >> 4;
  const __bf16* p = W + (size_t)(n0 + n) * Kdim + kb + half * 8;
  BFrag b;
  b.q[0] = *(const uint4*)(p);        // global_load_b128
  b.q[1] = *(const uint4*)(p + 16);   // global_load_b128
  return b.v;
}

__device__ inline v8f wmma_bf16(v16bf a, v16bf b, v8f c) {
  return __builtin_amdgcn_wmma_f32_16x16x32_bf16(false, a, false, b,
                                                 (short)0, c, false, false);
}

// ---------------- weight precision prep ----------------------------------

__global__ __launch_bounds__(256) void k_f32_to_bf16(const float* __restrict__ in,
                                                     __bf16* __restrict__ out, int n) {
  int i = blockIdx.x * 256 + threadIdx.x;
  if (i < n) out[i] = (__bf16)in[i];
}

// ---------------- batch statistics ---------------------------------------

__global__ __launch_bounds__(256) void k_stats_partial(const float* __restrict__ xin,
                                                       float* __restrict__ part) {
  __shared__ float ls[512];
  int tid = threadIdx.x;
  int c = tid & 63, rsub = tid >> 6;
  int r0 = blockIdx.x * 256;
  float s = 0.f, s2 = 0.f;
  for (int r = rsub; r < 256; r += 4) {
    float v = xin[(size_t)(r0 + r) * DD + c];
    s += v; s2 += v * v;
  }
  ls[tid] = s; ls[256 + tid] = s2;
  __syncthreads();
  if (tid < 64) {
    float ss  = ls[tid] + ls[tid + 64] + ls[tid + 128] + ls[tid + 192];
    float ss2 = ls[256 + tid] + ls[320 + tid] + ls[384 + tid] + ls[448 + tid];
    part[blockIdx.x * 128 + tid]      = ss;
    part[blockIdx.x * 128 + 64 + tid] = ss2;
  }
}

__global__ void k_stats_final(const float* __restrict__ part, float* __restrict__ stats) {
  int c = threadIdx.x;  // 64 threads
  float s = 0.f, s2 = 0.f;
  for (int b = 0; b < 256; ++b) { s += part[b * 128 + c]; s2 += part[b * 128 + 64 + c]; }
  float mean = s / (float)NN;
  float var  = s2 / (float)NN - mean * mean;
  stats[c]      = mean;
  stats[64 + c] = 1.0f / sqrtf(var + EPS);
}

// ---------------- fused coupling layer -----------------------------------

__global__ __launch_bounds__(256) void k_coupling(
    const float* __restrict__ xin, float* __restrict__ xout,
    const float* __restrict__ stats,
    const __bf16* __restrict__ sW1, const float* __restrict__ sb1,
    const __bf16* __restrict__ sW2, const float* __restrict__ sb2,
    const __bf16* __restrict__ sW3, const float* __restrict__ sb3,
    const __bf16* __restrict__ tW1, const float* __restrict__ tb1,
    const __bf16* __restrict__ tW2, const float* __restrict__ tb2,
    const __bf16* __restrict__ tW3, const float* __restrict__ tb3,
    float* __restrict__ det_s, int layer) {
  __shared__ __align__(16) float  s_stats[128];
  __shared__ __align__(16) __bf16 sA[TM * dd];   // x1 normalized, bf16
  __shared__ __align__(16) __bf16 sH [TM * HH];  // hidden 1 (64 KB)
  __shared__ __align__(16) __bf16 sH2[TM * HH];  // hidden 2 (64 KB)
  __shared__ __align__(16) float  s_os[TM * dd]; // s-MLP out (8 KB)
  __shared__ __align__(16) float  s_ot[TM * dd]; // t-MLP out (8 KB)

  const int tid  = threadIdx.x;
  const int wave = tid >> 5;
  const int lane = tid & 31;
  const int n    = lane & 15;
  const int half = lane >> 4;
  const int r0   = blockIdx.x * TM;

  if (tid < 128) s_stats[tid] = stats[tid];
  __syncthreads();

  const int x1off = (layer & 1) ? dd : 0;  // which cols form x1
  for (int idx = tid; idx < TM * dd; idx += 256) {
    int m = idx >> 5, j = idx & 31;
    int c1 = x1off + j;
    float v = xin[(size_t)(r0 + m) * DD + c1];
    sA[m * dd + j] = (__bf16)((v - s_stats[c1]) * s_stats[64 + c1]);
  }
  __syncthreads();

  // A fragments for stage 1 are invariant across nt and across both MLPs.
  v16bf aS1[4];
#pragma unroll
  for (int rt = 0; rt < 4; ++rt) aS1[rt] = load_a_lds16(sA + rt * 16 * dd, dd, lane);

  for (int which = 0; which < 2; ++which) {
    const __bf16* W1 = which ? tW1 : sW1; const float* b1 = which ? tb1 : sb1;
    const __bf16* W2 = which ? tW2 : sW2; const float* b2 = which ? tb2 : sb2;
    const __bf16* W3 = which ? tW3 : sW3; const float* b3 = which ? tb3 : sb3;
    float* out = which ? s_ot : s_os;

    // ---- stage 1: h = relu(x1(64x32) @ W1^T(32x512) + b1) -> sH
    for (int nt = wave; nt < HH / 16; nt += 8) {
      v16bf bfrag = load_b_w(W1, dd, nt * 16, 0, lane);
      float bias = b1[nt * 16 + n];
#pragma unroll
      for (int rt = 0; rt < 4; ++rt) {
        v8f acc = {};
        acc = wmma_bf16(aS1[rt], bfrag, acc);
        CFrag cf; cf.v = acc;
#pragma unroll
        for (int r = 0; r < 8; ++r) {
          int m = r + half * 8;
          float hv = cf.f[r] + bias;
          hv = hv > 0.f ? hv : 0.f;
          sH[(rt * 16 + m) * HH + nt * 16 + n] = (__bf16)hv;
        }
      }
    }
    __syncthreads();

    // ---- stage 2: h2 = relu(h(64x512) @ W2^T(512x512) + b2) -> sH2
    for (int nt = wave; nt < HH / 16; nt += 8) {
      v8f acc[4] = {};
      for (int kc = 0; kc < HH / 32; ++kc) {
        if (kc + 1 < HH / 32)  // warm next K-chunk (global_prefetch_b8)
          __builtin_prefetch(W2 + (size_t)(nt * 16 + n) * HH + (kc + 1) * 32, 0, 3);
        v16bf bfrag = load_b_w(W2, HH, nt * 16, kc * 32, lane);
#pragma unroll
        for (int rt = 0; rt < 4; ++rt) {
          v16bf a = load_a_lds16(sH + (rt * 16) * HH + kc * 32, HH, lane);
          acc[rt] = wmma_bf16(a, bfrag, acc[rt]);
        }
      }
      float bias = b2[nt * 16 + n];
#pragma unroll
      for (int rt = 0; rt < 4; ++rt) {
        CFrag cf; cf.v = acc[rt];
#pragma unroll
        for (int r = 0; r < 8; ++r) {
          int m = r + half * 8;
          float hv = cf.f[r] + bias;
          hv = hv > 0.f ? hv : 0.f;
          sH2[(rt * 16 + m) * HH + nt * 16 + n] = (__bf16)hv;
        }
      }
    }
    __syncthreads();

    // ---- stage 3: out = h2(64x512) @ W3^T(512x32) + b3 ; 8 jobs = 8 waves
    {
      int nt = wave & 1, rt = wave >> 1;
      v8f acc = {};
      for (int kc = 0; kc < HH / 32; ++kc) {
        v16bf bfrag = load_b_w(W3, HH, nt * 16, kc * 32, lane);
        v16bf a = load_a_lds16(sH2 + (rt * 16) * HH + kc * 32, HH, lane);
        acc = wmma_bf16(a, bfrag, acc);
      }
      float bias = b3[nt * 16 + n];
      CFrag cf; cf.v = acc;
#pragma unroll
      for (int r = 0; r < 8; ++r) {
        int m = r + half * 8;
        out[(rt * 16 + m) * dd + nt * 16 + n] = cf.f[r] + bias;
      }
    }
    __syncthreads();
  }

  // ---- affine coupling: recompute normalized x1/x2 (sH is free -> f32 scratch)
  float* fx1 = (float*)sH;
  float* fx2 = (float*)sH + TM * dd;
  for (int idx = tid; idx < TM * dd; idx += 256) {
    int m = idx >> 5, j = idx & 31;
    size_t R = (size_t)(r0 + m);
    int c1 = x1off + j;
    int c2 = (x1off == 0) ? (dd + j) : j;
    fx1[idx] = (xin[R * DD + c1] - s_stats[c1]) * s_stats[64 + c1];
    fx2[idx] = (xin[R * DD + c2] - s_stats[c2]) * s_stats[64 + c2];
  }
  __syncthreads();  // all reads of xin done before any write of xout (aliased)
  for (int idx = tid; idx < TM * dd; idx += 256) {
    int m = idx >> 5, j = idx & 31;
    size_t R = (size_t)(r0 + m);
    float y2 = fx2[idx] * expf(s_os[idx]) + s_ot[idx];
    xout[R * DD + j]      = fx1[idx];
    xout[R * DD + dd + j] = y2;
  }
  if (tid < TM) {
    float ds = 0.f;
    for (int j = 0; j < dd; ++j) ds += s_os[tid * dd + j];
    size_t R = (size_t)(r0 + tid);
    det_s[R] = (layer == 0) ? ds : det_s[R] + ds;
  }
}

// ---------------- final likelihood + reductions ---------------------------

__global__ __launch_bounds__(256) void k_final_partial(const float* __restrict__ xbuf,
                                                       const float* __restrict__ det_s,
                                                       float* __restrict__ out,
                                                       float* __restrict__ part) {
  __shared__ float red[512];
  int tid = threadIdx.x;
  size_t R = (size_t)blockIdx.x * 256 + tid;  // 256 blocks x 256 rows
  float ss = 0.f;
  for (int c = 0; c < DD; ++c) {
    float v = xbuf[R * DD + c];
    out[R * DD + c] = v;
    ss += v * v;
  }
  const float LOG2PI = 1.8378770664093453f;
  float ll = -0.5f * ss - 0.5f * (float)DD * LOG2PI;
  float ds = det_s[R];
  out[(size_t)NN * DD + 1 + R]      = ll;
  out[(size_t)NN * DD + 1 + NN + R] = ds;
  red[tid] = ll; red[256 + tid] = ds;
  __syncthreads();
  for (int s = 128; s > 0; s >>= 1) {
    if (tid < s) { red[tid] += red[tid + s]; red[256 + tid] += red[256 + tid + s]; }
    __syncthreads();
  }
  if (tid == 0) { part[blockIdx.x * 2] = red[0]; part[blockIdx.x * 2 + 1] = red[256]; }
}

__global__ __launch_bounds__(256) void k_final_reduce(const float* __restrict__ part,
                                                      float* __restrict__ out) {
  __shared__ float red[512];
  int tid = threadIdx.x;
  red[tid] = part[tid * 2]; red[256 + tid] = part[tid * 2 + 1];
  __syncthreads();
  for (int s = 128; s > 0; s >>= 1) {
    if (tid < s) { red[tid] += red[tid + s]; red[256 + tid] += red[256 + tid + s]; }
    __syncthreads();
  }
  if (tid == 0) {
    float ll_mean = red[0] / (float)NN;
    float ds_mean = red[256] / (float)NN;
    out[(size_t)NN * DD]                  = -(ds_mean + ll_mean);  // loss
    out[(size_t)NN * DD + 1 + 2 * NN]     = ll_mean;
    out[(size_t)NN * DD + 1 + 2 * NN + 1] = ds_mean;
  }
}

// ---------------- host orchestration --------------------------------------

extern "C" void kernel_launch(void* const* d_in, const int* in_sizes, int n_in,
                              void* d_out, int out_size, void* d_ws, size_t ws_size,
                              hipStream_t stream) {
  (void)in_sizes; (void)n_in; (void)out_size; (void)ws_size;
  const float* x   = (const float*)d_in[0];
  const float* sW1 = (const float*)d_in[1];  const float* sb1 = (const float*)d_in[2];
  const float* sW2 = (const float*)d_in[3];  const float* sb2 = (const float*)d_in[4];
  const float* sW3 = (const float*)d_in[5];  const float* sb3 = (const float*)d_in[6];
  const float* tW1 = (const float*)d_in[7];  const float* tb1 = (const float*)d_in[8];
  const float* tW2 = (const float*)d_in[9];  const float* tb2 = (const float*)d_in[10];
  const float* tW3 = (const float*)d_in[11]; const float* tb3 = (const float*)d_in[12];

  char* ws = (char*)d_ws;
  size_t off = 0;
  float* xbuf  = (float*)(ws + off); off += (size_t)NN * DD * 4;   // 16 MB
  float* det   = (float*)(ws + off); off += (size_t)NN * 4;        // 256 KB
  float* stats = (float*)(ws + off); off += 128 * 4;
  float* part  = (float*)(ws + off); off += 256 * 128 * 4;
  off = (off + 255) & ~(size_t)255;
  __bf16* bsW1 = (__bf16*)(ws + off); off += (size_t)LAYERS * W1_SZ * 2;
  __bf16* bsW2 = (__bf16*)(ws + off); off += (size_t)LAYERS * W2_SZ * 2;
  __bf16* bsW3 = (__bf16*)(ws + off); off += (size_t)LAYERS * W3_SZ * 2;
  __bf16* btW1 = (__bf16*)(ws + off); off += (size_t)LAYERS * W1_SZ * 2;
  __bf16* btW2 = (__bf16*)(ws + off); off += (size_t)LAYERS * W2_SZ * 2;
  __bf16* btW3 = (__bf16*)(ws + off); off += (size_t)LAYERS * W3_SZ * 2;

  int n;
  n = LAYERS * W1_SZ; k_f32_to_bf16<<<(n + 255) / 256, 256, 0, stream>>>(sW1, bsW1, n);
  n = LAYERS * W2_SZ; k_f32_to_bf16<<<(n + 255) / 256, 256, 0, stream>>>(sW2, bsW2, n);
  n = LAYERS * W3_SZ; k_f32_to_bf16<<<(n + 255) / 256, 256, 0, stream>>>(sW3, bsW3, n);
  n = LAYERS * W1_SZ; k_f32_to_bf16<<<(n + 255) / 256, 256, 0, stream>>>(tW1, btW1, n);
  n = LAYERS * W2_SZ; k_f32_to_bf16<<<(n + 255) / 256, 256, 0, stream>>>(tW2, btW2, n);
  n = LAYERS * W3_SZ; k_f32_to_bf16<<<(n + 255) / 256, 256, 0, stream>>>(tW3, btW3, n);

  for (int L = 0; L < LAYERS; ++L) {
    const float* xi = (L == 0) ? x : xbuf;
    k_stats_partial<<<256, 256, 0, stream>>>(xi, part);
    k_stats_final<<<1, 64, 0, stream>>>(part, stats);
    k_coupling<<<NN / TM, 256, 0, stream>>>(
        xi, xbuf, stats,
        bsW1 + (size_t)L * W1_SZ, sb1 + (size_t)L * B1_SZ,
        bsW2 + (size_t)L * W2_SZ, sb2 + (size_t)L * B2_SZ,
        bsW3 + (size_t)L * W3_SZ, sb3 + (size_t)L * B3_SZ,
        btW1 + (size_t)L * W1_SZ, tb1 + (size_t)L * B1_SZ,
        btW2 + (size_t)L * W2_SZ, tb2 + (size_t)L * B2_SZ,
        btW3 + (size_t)L * W3_SZ, tb3 + (size_t)L * B3_SZ,
        det, L);
  }
  k_final_partial<<<256, 256, 0, stream>>>(xbuf, det, (float*)d_out, part);
  k_final_reduce<<<1, 256, 0, stream>>>(part, (float*)d_out);
}
